// Network_3908420240174
// MI455X (gfx1250) — matrix operand order repaired and checked by reading, not compiled, not run on hardware
//
#include <hip/hip_runtime.h>

typedef __attribute__((ext_vector_type(16))) _Float16 v16h;
typedef __attribute__((ext_vector_type(8)))  _Float16 v8h;
typedef __attribute__((ext_vector_type(8)))  float    v8f;

#define BATCH 64
#define SSTRIDE 256   // channel stride (halves) of the f16 spike staging buffers

// ---------------------------------------------------------------------------
// Transpose input spikes [B,20,T] f32  ->  sA [b][t][c] f16, c padded 20->32
// (pad rows t in [T,Tpad) are left as-is; they only feed unread z pad rows)
// ---------------------------------------------------------------------------
__global__ void snn_xpose_in(const float* __restrict__ in, _Float16* __restrict__ sA,
                             int T, int Tpad) {
  int idx = blockIdx.x * blockDim.x + threadIdx.x;
  int total = BATCH * T * 32;
  if (idx >= total) return;
  int c = idx & 31;
  int t = (idx >> 5) % T;
  int b = idx / (32 * T);
  float v = (c < 20) ? in[((size_t)b * 20 + c) * T + t] : 0.0f;
  sA[((size_t)b * Tpad + t) * SSTRIDE + c] = (_Float16)v;
}

// ---------------------------------------------------------------------------
// Convert weights [Cout,Cin] f32 -> wbuf [Opad][Cpad] f16, zero padded
// ---------------------------------------------------------------------------
__global__ void snn_cvt_w(const float* __restrict__ W, _Float16* __restrict__ wbuf,
                          int Cout, int Cin, int Cpad, int Opad) {
  int idx = blockIdx.x * blockDim.x + threadIdx.x;
  int total = Opad * Cpad;
  if (idx >= total) return;
  int o = idx / Cpad, c = idx % Cpad;
  float v = (o < Cout && c < Cin) ? W[o * Cin + c] : 0.0f;
  wbuf[idx] = (_Float16)v;
}

// ---------------------------------------------------------------------------
// GEMM: z[b][t][o] = sum_c sPrev[b][t][c] * wbuf[o][c]
// One wave32 per 16(t) x 32(o) tile pair: one A fragment feeds two WMMAs.
// All loads/stores unguarded (time padded to Tpad), so EXEC stays all-1s and
// the inner loop is pure loads + v_wmma_f32_16x16x32_f16.
// ---------------------------------------------------------------------------
__global__ void snn_gemm_f16(const _Float16* __restrict__ sPrev,   // [B*Tpad, SSTRIDE]
                             const _Float16* __restrict__ wbuf,    // [Opad, Cpad]
                             float* __restrict__ z,                // [B*Tpad, Opad]
                             int Tpad, int Tt, int Cpad, int Opad, int totalTiles) {
  int wid  = threadIdx.x >> 5;
  int lane = threadIdx.x & 31;
  int tIdx = blockIdx.x * (blockDim.x >> 5) + wid;
  if (tIdx >= totalTiles) return;           // wave-uniform

  int oPairs = Opad >> 5;
  int oPair  = tIdx % oPairs;
  int rest   = tIdx / oPairs;
  int tTile  = rest % Tt;
  int b      = rest / Tt;
  int tBase  = tTile * 16;
  int oBase  = oPair * 32;

  int half = lane >> 4;     // 0: lanes 0-15, 1: lanes 16-31
  int mn   = lane & 15;     // A row (t_local) / B col (o_local) / C col

  // A: lane holds row M=mn; halves h -> K = kc + 16*half + h (contiguous)
  const _Float16* aRow  = sPrev + ((size_t)b * Tpad + tBase + mn) * SSTRIDE + half * 16;
  // B: lane holds column N=mn; halves h -> K = kc + 16*half + h (contiguous)
  const _Float16* b0Col = wbuf + (size_t)(oBase + mn) * Cpad + half * 16;
  const _Float16* b1Col = b0Col + (size_t)16 * Cpad;

  v8f acc0 = {};
  v8f acc1 = {};
  for (int kc = 0; kc < Cpad; kc += 32) {
    union { v16h v; v8h h[2]; } A;
    A.h[0] = *(const v8h*)(aRow + kc);
    A.h[1] = *(const v8h*)(aRow + kc + 8);
    v16h B0 = *(const v16h*)(b0Col + kc);
    v16h B1 = *(const v16h*)(b1Col + kc);
    acc0 = __builtin_amdgcn_wmma_f32_16x16x32_f16(
        false, A.v, false, B0, (short)0, acc0, false, false);
    acc1 = __builtin_amdgcn_wmma_f32_16x16x32_f16(
        false, A.v, false, B1, (short)0, acc1, false, false);
  }

  // C/D: lane holds N=mn, rows M = 8*half + v
  float* zRow = z + ((size_t)b * Tpad + tBase + half * 8) * Opad + oBase + mn;
#pragma unroll
  for (int v = 0; v < 8; ++v) {
    zRow[(size_t)v * Opad]      = acc0[v];
    zRow[(size_t)v * Opad + 16] = acc1[v];
  }
}

// ---------------------------------------------------------------------------
// CUBA-LIF scan over T per (b, o).  Writes f16 spikes for next layer and/or
// f32 spikes [B][Cout][T] to d_out for the final layer; accumulates count.
// ---------------------------------------------------------------------------
__global__ void snn_scan(const float* __restrict__ z, _Float16* __restrict__ sNext,
                         float* __restrict__ outF, float* __restrict__ countAcc,
                         int T, int Tpad, int Cout, int Opad) {
  int idx = blockIdx.x * blockDim.x + threadIdx.x;
  int total = BATCH * Cout;
  if (idx >= total) return;
  int b = idx / Cout;
  int o = idx % Cout;
  const float* zp = z + (size_t)b * Tpad * Opad + o;
  _Float16* sp = sNext ? (sNext + (size_t)b * Tpad * SSTRIDE + o) : nullptr;
  float*    op = outF  ? (outF + ((size_t)b * Cout + o) * T)      : nullptr;

  float cur = 0.0f, volt = 0.0f, cnt = 0.0f;
  for (int t = 0; t < T; ++t) {
    cur  = 0.75f * cur + zp[(size_t)t * Opad];   // (1 - CURRENT_DECAY)
    volt = 0.97f * volt + cur;                   // (1 - VOLTAGE_DECAY)
    float s = (volt >= 1.25f) ? 1.0f : 0.0f;     // THRESHOLD
    volt = (s > 0.0f) ? 0.0f : volt;             // reset on spike
    cnt += s;
    if (sp) sp[(size_t)t * SSTRIDE] = (_Float16)s;
    if (op) op[t] = s;
  }
  atomicAdd(countAcc, cnt);
}

__global__ void snn_zero4(float* p) {
  if (threadIdx.x < 4 && blockIdx.x == 0) p[threadIdx.x] = 0.0f;
}

__global__ void snn_final(const float* __restrict__ counts, float* __restrict__ out, int T) {
  if (threadIdx.x < 4 && blockIdx.x == 0) {
    float denom = (threadIdx.x == 3) ? (float)BATCH * 35.0f * (float)T
                                     : (float)BATCH * 256.0f * (float)T;
    out[threadIdx.x] = counts[threadIdx.x] / denom;
  }
}

// ---------------------------------------------------------------------------
extern "C" void kernel_launch(void* const* d_in, const int* in_sizes, int n_in,
                              void* d_out, int out_size, void* d_ws, size_t ws_size,
                              hipStream_t stream) {
  const float* spike = (const float*)d_in[0];
  const float* W[4] = { (const float*)d_in[1], (const float*)d_in[2],
                        (const float*)d_in[3], (const float*)d_in[4] };
  float* out = (float*)d_out;

  const int T    = in_sizes[0] / (BATCH * 20);   // 1000
  const int Tt   = (T + 15) / 16;
  const int Tpad = Tt * 16;                      // 1008

  // Workspace carve-up
  char* ws = (char*)d_ws;
  const size_t sBytes = (size_t)BATCH * Tpad * SSTRIDE * sizeof(_Float16);
  _Float16* sA   = (_Float16*)(ws);
  _Float16* sB   = (_Float16*)(ws + sBytes);
  float*    z    = (float*)   (ws + 2 * sBytes);                       // B*Tpad*256 f32
  _Float16* wbuf = (_Float16*)(ws + 2 * sBytes + (size_t)BATCH * Tpad * 256 * sizeof(float));
  float*    cnts = (float*)((char*)wbuf + 256 * 256 * sizeof(_Float16));

  snn_zero4<<<1, 32, 0, stream>>>(cnts);

  // Stage input spikes as f16 [b][t][c], c padded to 32
  {
    int total = BATCH * T * 32;
    snn_xpose_in<<<(total + 255) / 256, 256, 0, stream>>>(spike, sA, T, Tpad);
  }

  struct Layer { int Cin, Cout, Cpad, Opad; };
  const Layer L[4] = { {20, 256, 32, 256}, {256, 256, 256, 256},
                       {256, 256, 256, 256}, {256, 35, 256, 64} };

  _Float16* sIn = sA;
  _Float16* sOut = sB;
  for (int l = 0; l < 4; ++l) {
    // Weights -> padded f16
    {
      int total = L[l].Opad * L[l].Cpad;
      snn_cvt_w<<<(total + 255) / 256, 256, 0, stream>>>(W[l], wbuf, L[l].Cout,
                                                         L[l].Cin, L[l].Cpad, L[l].Opad);
    }
    // GEMM: one wave per 16x32 tile pair, 8 waves per block
    {
      int totalTiles = BATCH * Tt * (L[l].Opad >> 5);
      int blocks = (totalTiles + 7) / 8;
      snn_gemm_f16<<<blocks, 256, 0, stream>>>(sIn, wbuf, z, Tpad, Tt,
                                               L[l].Cpad, L[l].Opad, totalTiles);
    }
    // LIF scan
    {
      int total = BATCH * L[l].Cout;
      bool last = (l == 3);
      snn_scan<<<(total + 255) / 256, 256, 0, stream>>>(
          z, last ? (_Float16*)nullptr : sOut, last ? out : (float*)nullptr,
          cnts + l, T, Tpad, L[l].Cout, L[l].Opad);
    }
    _Float16* tmp = sIn; sIn = sOut; sOut = tmp;
  }

  // Append the 4 per-layer means after the [B,35,T] spikes
  snn_final<<<1, 32, 0, stream>>>(cnts, out + (size_t)BATCH * 35 * T, T);
}
